// FlashAttention_24026047054195
// MI455X (gfx1250) — matrix operand order repaired and checked by reading, not compiled, not run on hardware
//
#include <hip/hip_runtime.h>
#include <hip/hip_bf16.h>

// ---------------------------------------------------------------------------
// MHA forward for B=1, N=4096, C=1024, H=16, D=64 on gfx1250 (CDNA5, wave32)
// GEMMs: workgroup-cooperative, B-panel TDM-loaded into double-buffered LDS.
// Attention: per-wave flash attention, K/V tiles TDM-loaded into LDS.
// All matmuls via v_wmma_f32_16x16x32_f16.
// ---------------------------------------------------------------------------

typedef __attribute__((ext_vector_type(16))) _Float16 v16h;
typedef __attribute__((ext_vector_type(8)))  float    v8f;
typedef __attribute__((ext_vector_type(4)))  unsigned int u32x4;
typedef __attribute__((ext_vector_type(8)))  int      i32x8;

#define NTOK  4096
#define CDIM  1024
#define HEADS 16
#define DH    64

union Fr16 { u32x4 q[2]; v16h h; };

static __device__ inline v8f vzero8() {
  v8f z = {0.f, 0.f, 0.f, 0.f, 0.f, 0.f, 0.f, 0.f};
  return z;
}

static __device__ inline v8f wmma_f16(const Fr16& a, const Fr16& b, v8f c) {
  // (neg_a, A, neg_b, B, c_mod, C, reuse_a, reuse_b)
  return __builtin_amdgcn_wmma_f32_16x16x32_f16(false, a.h, false, b.h,
                                                (short)0, c, false, false);
}

// A-fragment: 16 halves at p[0..7] and p[16..23]
static __device__ inline void load_afrag(Fr16& a, const _Float16* p) {
  a.q[0] = *(const u32x4*)(p);
  a.q[1] = *(const u32x4*)(p + 16);
}
// B-fragment: 16 halves at p[0..7] and p[16..23]
static __device__ inline void load_bfrag(Fr16& b, const _Float16* p) {
  b.q[0] = *(const u32x4*)(p);
  b.q[1] = *(const u32x4*)(p + 16);
}

// ---------------------------------------------------------------------------
// Tensor Data Mover: 2D tile load Global -> LDS (f16 elements).
// D# descriptor (group0 128b + group1 256b) per CDNA5 ISA section 8.3/8.4.
// ---------------------------------------------------------------------------
static __device__ inline void tdm_load_2d(unsigned lds_off, const void* gp,
                                          unsigned tdim0, unsigned tdim1,
                                          unsigned tild0, unsigned tild1,
                                          unsigned stride0) {
  unsigned long long ga = (unsigned long long)(size_t)gp;
  u32x4 g0;
  g0[0] = 1u;                                   // count=1, user descriptor
  g0[1] = lds_off;                              // lds_addr (bytes)
  g0[2] = (unsigned)ga;                         // global_addr[31:0]
  g0[3] = (unsigned)((ga >> 32) & 0x01FFFFFFu)  // global_addr[56:32]
        | (2u << 30);                           // type = 2 ("image")
  i32x8 g1;
  g1[0] = (int)(1u << 16);                      // wg_mask=0, data_size=1 (2B)
  g1[1] = (int)((tdim0 & 0xFFFFu) << 16);       // tensor_dim0[15:0]
  g1[2] = (int)(((tdim0 >> 16) & 0xFFFFu) | ((tdim1 & 0xFFFFu) << 16));
  g1[3] = (int)(((tdim1 >> 16) & 0xFFFFu) | ((tild0 & 0xFFFFu) << 16));
  g1[4] = (int)(tild1 & 0xFFFFu);               // tile_dim1, tile_dim2=0
  g1[5] = (int)stride0;                         // tensor_dim0_stride[31:0]
  g1[6] = 0;
  g1[7] = 0;
  asm volatile("tensor_load_to_lds %0, %1" :: "s"(g0), "s"(g1) : "memory");
}

// Low 32 bits of an aperture-based generic pointer to LDS = LDS byte offset.
static __device__ inline unsigned lds_off_of(const void* p) {
  return (unsigned)(size_t)p;
}

// ---------------------------------------------------------------------------
// f32 -> f16 elementwise convert
// ---------------------------------------------------------------------------
__global__ __launch_bounds__(256) void cvt_f16_kernel(
    const float* __restrict__ in, _Float16* __restrict__ out, int n) {
  int i = blockIdx.x * blockDim.x + threadIdx.x;
  if (i < n) out[i] = (_Float16)in[i];
}

// f32 [rows][cols] -> f16 transposed [cols][rows]
__global__ __launch_bounds__(256) void cvt_t_f16_kernel(
    const float* __restrict__ in, _Float16* __restrict__ out, int rows, int cols) {
  int i = blockIdx.x * blockDim.x + threadIdx.x;
  if (i < rows * cols) {
    int r = i / cols;
    int c = i - r * cols;
    out[(size_t)c * rows + r] = (_Float16)in[i];
  }
}

// ---------------------------------------------------------------------------
// Cooperative GEMM core: block of 4 waves computes 64(M) x 64(N).
// Wave w owns rows m0 = blockRow*64 + w*16; all waves share the B^T k-panel
// (64 cols x 64 k, 8 KB f16) which wave 0 TDM-loads into double-buffered LDS.
// Sync: wave0 s_wait_tensorcnt, then workgroup barrier; second barrier before
// the buffer is overwritten. A-fragments are per-wave global b128 loads.
// Requires kdim % 64 == 0.
// ---------------------------------------------------------------------------
static __device__ inline void mma_tile_coop(const _Float16* __restrict__ A, int lda,
                                            const _Float16* __restrict__ Bt, int ldb,
                                            int m0, int n0, int kdim,
                                            _Float16* bsh /*2*64*64 halves LDS*/,
                                            v8f acc[4]) {
  const int wave = threadIdx.x >> 5;
  const int lane = threadIdx.x & 31;
  const int lg = lane & 15;
  const int kh = lane >> 4;
  const _Float16* arow = A + (size_t)(m0 + lg) * lda + kh * 8;
  const _Float16* bgl = Bt + (size_t)n0 * ldb;

  if (wave == 0)
    tdm_load_2d(lds_off_of(bsh), bgl, (unsigned)kdim, 64u, 64u, 64u, (unsigned)ldb);

  const int iters = kdim >> 6;
  for (int it = 0; it < iters; ++it) {
    const int k0 = it << 6;
    const unsigned cofs = (it & 1) ? 4096u : 0u;     // current buffer (halves)
    if (wave == 0) {
      if (it + 1 < iters) {
        tdm_load_2d(lds_off_of(bsh + (cofs ^ 4096u)), bgl + k0 + 64,
                    (unsigned)kdim, 64u, 64u, 64u, (unsigned)ldb);
        asm volatile("s_wait_tensorcnt 0x1" ::: "memory");
      } else {
        asm volatile("s_wait_tensorcnt 0x0" ::: "memory");
      }
    }
    __syncthreads();                                  // panel ready for all waves

    Fr16 a0, a1;
    load_afrag(a0, arow + k0);
    load_afrag(a1, arow + k0 + 32);
    const _Float16* bl = bsh + cofs + (size_t)lg * 64 + kh * 8;
#pragma unroll
    for (int t = 0; t < 4; ++t) {
      Fr16 bb0, bb1;
      load_bfrag(bb0, bl + (size_t)(t * 16) * 64);
      load_bfrag(bb1, bl + (size_t)(t * 16) * 64 + 32);
      acc[t] = wmma_f16(a0, bb0, acc[t]);
      acc[t] = wmma_f16(a1, bb1, acc[t]);
    }
    __syncthreads();                                  // all reads done before overwrite
  }
}

// ---------------------------------------------------------------------------
// QKV projection: xh[4096][1024] @ Wqkv_t[3072][1024]^T + b_qkv
//   three==0 -> Q[h][tok][d] (pre-scaled by 1/8)
//   three==1 -> K[h][tok][d]
//   three==2 -> VT[h][d][tok]
// Grid: 64 row-blocks x 48 col-blocks = 3072 blocks of 128 threads.
// ---------------------------------------------------------------------------
__global__ __launch_bounds__(128) void qkv_gemm_kernel(
    const _Float16* __restrict__ xh, const _Float16* __restrict__ wt,
    const float* __restrict__ bias,
    _Float16* __restrict__ Qd, _Float16* __restrict__ Kd, _Float16* __restrict__ VTd) {
  __shared__ __align__(16) _Float16 bsh[2 * 64 * 64];   // 16 KB

  const int wave = threadIdx.x >> 5;
  const int rb = blockIdx.x & 63;
  const int cb = blockIdx.x >> 6;          // 0..47
  const int m0 = rb * 64 + wave * 16;
  const int n0 = cb * 64;

  v8f acc[4] = {vzero8(), vzero8(), vzero8(), vzero8()};
  mma_tile_coop(xh, CDIM, wt, CDIM, m0, n0, CDIM, bsh, acc);

  const int lane = threadIdx.x & 31;
  const int lg = lane & 15;
  const int kh = lane >> 4;
#pragma unroll
  for (int t = 0; t < 4; ++t) {
    int c = n0 + t * 16 + lg;
    float bv = bias[c];
    int three = c >> 10;
    int r = c & 1023;
    int h = r >> 6;
    int d = r & 63;
#pragma unroll
    for (int j = 0; j < 8; ++j) {
      int token = m0 + j + (kh << 3);
      float v = acc[t][j] + bv;
      if (three == 0) {
        Qd[((size_t)h * NTOK + token) * DH + d] = (_Float16)(v * 0.125f);
      } else if (three == 1) {
        Kd[((size_t)h * NTOK + token) * DH + d] = (_Float16)v;
      } else {
        VTd[((size_t)h * DH + d) * NTOK + token] = (_Float16)v;
      }
    }
  }
}

// ---------------------------------------------------------------------------
// Flash attention: one wave owns 16 query rows of one head; streams 32-key
// blocks. K (32x64) and V^T (64x32) tiles TDM-loaded into per-wave LDS double
// buffers; next block's DMAs in flight during current block's WMMA + softmax.
// All LDS addressing via integer offsets from a single per-wave base so the
// accesses lower to ds_load/ds_store (not flat).
// ---------------------------------------------------------------------------
__global__ __launch_bounds__(128) void attn_kernel(
    const _Float16* __restrict__ Qd, const _Float16* __restrict__ Kd,
    const _Float16* __restrict__ VTd, _Float16* __restrict__ Od) {
  // Per-wave halves: K 2x2048, V 2x2048, P 512 = 8704 (17 KB); 4 waves = 68 KB.
  __shared__ __align__(16) _Float16 smem[4][8704];

  const int wave = threadIdx.x >> 5;
  const int lane = threadIdx.x & 31;
  const int lg = lane & 15;
  const int kh = lane >> 4;

  int gw = blockIdx.x * 4 + wave;     // 0..4095
  int head = gw >> 8;
  int q0 = (gw & 255) << 4;

  const _Float16* Qh = Qd + (size_t)head * NTOK * DH;
  const _Float16* Kh = Kd + (size_t)head * NTOK * DH;
  const _Float16* Vh = VTd + (size_t)head * DH * NTOK;

  _Float16* wb = &smem[wave][0];      // single LDS base; offsets select buffers

  // Q A-fragments (two K=32 chunks of D=64), held for the whole pass
  const _Float16* qrow = Qh + (size_t)(q0 + lg) * DH + kh * 8;
  Fr16 qa0, qa1;
  qa0.q[0] = *(const u32x4*)(qrow);
  qa0.q[1] = *(const u32x4*)(qrow + 16);
  qa1.q[0] = *(const u32x4*)(qrow + 32);
  qa1.q[1] = *(const u32x4*)(qrow + 48);

  v8f oo[4] = {vzero8(), vzero8(), vzero8(), vzero8()};
  float m[8], l[8];
#pragma unroll
  for (int j = 0; j < 8; ++j) { m[j] = -1e30f; l[j] = 0.f; }

  // Prologue: DMA block 0 into buffer 0.
  // K tile: 32 token-rows (stride 64), width 64 d   -> LDS [32][64]
  // V tile: 64 dim-rows  (stride 4096), width 32 tok -> LDS [64][32]
  tdm_load_2d(lds_off_of(wb), Kh, DH, NTOK, DH, 32, DH);
  tdm_load_2d(lds_off_of(wb + 4096), Vh, NTOK, DH, 32, DH, NTOK);

  for (int kb = 0; kb < NTOK; kb += 32) {
    const unsigned cofs = ((kb >> 5) & 1) ? 2048u : 0u;

    if (kb + 32 < NTOK) {
      tdm_load_2d(lds_off_of(wb + (cofs ^ 2048u)), Kh + (size_t)(kb + 32) * DH,
                  DH, NTOK, DH, 32, DH);
      tdm_load_2d(lds_off_of(wb + 4096 + (cofs ^ 2048u)), Vh + (kb + 32),
                  NTOK, DH, 32, DH, NTOK);
      asm volatile("s_wait_tensorcnt 0x2" ::: "memory");
    } else {
      asm volatile("s_wait_tensorcnt 0x0" ::: "memory");
    }

    // K B-fragments from LDS: 2 key tiles x 2 inner (D) chunks
    const _Float16* kp0 = wb + cofs + (size_t)lg * DH + kh * 8;
    const _Float16* kp1 = kp0 + 16 * DH;
    Fr16 b00, b01, b10, b11;
    load_bfrag(b00, kp0);
    load_bfrag(b01, kp0 + 32);
    load_bfrag(b10, kp1);
    load_bfrag(b11, kp1 + 32);

    v8f s0 = vzero8(), s1 = vzero8();
    s0 = wmma_f16(qa0, b00, s0);
    s0 = wmma_f16(qa1, b01, s0);
    s1 = wmma_f16(qa0, b10, s1);
    s1 = wmma_f16(qa1, b11, s1);

    // Online softmax per row (row = j + 8*kh, columns across 16-lane group)
#pragma unroll
    for (int j = 0; j < 8; ++j) {
      float a = s0[j];
      float b = s1[j];
      float mx = fmaxf(a, b);
      mx = fmaxf(mx, __shfl_xor(mx, 1));
      mx = fmaxf(mx, __shfl_xor(mx, 2));
      mx = fmaxf(mx, __shfl_xor(mx, 4));
      mx = fmaxf(mx, __shfl_xor(mx, 8));
      float mn = fmaxf(m[j], mx);
      float alpha = __expf(m[j] - mn);
      float p0 = __expf(a - mn);
      float p1 = __expf(b - mn);
      float rs = p0 + p1;
      rs += __shfl_xor(rs, 1);
      rs += __shfl_xor(rs, 2);
      rs += __shfl_xor(rs, 4);
      rs += __shfl_xor(rs, 8);
      l[j] = l[j] * alpha + rs;
      m[j] = mn;
      oo[0][j] *= alpha;
      oo[1][j] *= alpha;
      oo[2][j] *= alpha;
      oo[3][j] *= alpha;
      // Stage P (C-layout) into LDS for A-layout reload
      int r = j + (kh << 3);
      wb[8192 + r * 32 + lg]      = (_Float16)p0;
      wb[8192 + r * 32 + 16 + lg] = (_Float16)p1;
    }

    // P A-fragment (16x32) back from LDS (DS ops in-order within a wave)
    const _Float16* prd = wb + 8192 + lg * 32 + kh * 8;
    Fr16 pa;
    pa.q[0] = *(const u32x4*)(prd);
    pa.q[1] = *(const u32x4*)(prd + 16);

    // P @ V: 4 output d-tiles, inner K = 32 keys, V^T tile in LDS [64][32]
    const _Float16* vp = wb + 4096 + cofs + (size_t)lg * 32 + kh * 8;
#pragma unroll
    for (int t = 0; t < 4; ++t) {
      Fr16 vb;
      load_bfrag(vb, vp + (size_t)(t * 16) * 32);
      oo[t] = wmma_f16(pa, vb, oo[t]);
    }
  }

  // Normalize by l and store f16 in [token][h*64 + d] layout
#pragma unroll
  for (int j = 0; j < 8; ++j) {
    float inv = 1.0f / l[j];
    int token = q0 + j + (kh << 3);
    size_t base = (size_t)token * CDIM + head * DH + lg;
    Od[base]      = (_Float16)(oo[0][j] * inv);
    Od[base + 16] = (_Float16)(oo[1][j] * inv);
    Od[base + 32] = (_Float16)(oo[2][j] * inv);
    Od[base + 48] = (_Float16)(oo[3][j] * inv);
  }
}

// ---------------------------------------------------------------------------
// Output projection: attn[4096][1024] @ Wout_t[1024][1024]^T + b_out -> f32
// Grid: 64 row-blocks x 16 col-blocks = 1024 blocks of 128 threads.
// ---------------------------------------------------------------------------
__global__ __launch_bounds__(128) void out_gemm_kernel(
    const _Float16* __restrict__ ah, const _Float16* __restrict__ wt,
    const float* __restrict__ bias, float* __restrict__ out) {
  __shared__ __align__(16) _Float16 bsh[2 * 64 * 64];   // 16 KB

  const int wave = threadIdx.x >> 5;
  const int rb = blockIdx.x & 63;
  const int cb = blockIdx.x >> 6;          // 0..15
  const int m0 = rb * 64 + wave * 16;
  const int n0 = cb * 64;

  v8f acc[4] = {vzero8(), vzero8(), vzero8(), vzero8()};
  mma_tile_coop(ah, CDIM, wt, CDIM, m0, n0, CDIM, bsh, acc);

  const int lane = threadIdx.x & 31;
  const int lg = lane & 15;
  const int kh = lane >> 4;
#pragma unroll
  for (int t = 0; t < 4; ++t) {
    int c = n0 + t * 16 + lg;
    float bv = bias[c];
#pragma unroll
    for (int j = 0; j < 8; ++j) {
      int token = m0 + j + (kh << 3);
      out[(size_t)token * CDIM + c] = acc[t][j] + bv;
    }
  }
}

// ---------------------------------------------------------------------------
// Host launcher
// Workspace layout (bytes):
//   [0,   8M)  xh  : x as f16            (reused as attention output)
//   [8M, 14M)  wqt : W_qkv^T f16 [3072][1024]
//   [14M,16M)  wot : W_out^T f16 [1024][1024]
//   [16M,24M)  Q f16 [16][4096][64]
//   [24M,32M)  K f16 [16][4096][64]
//   [32M,40M)  V^T f16 [16][64][4096]
// ---------------------------------------------------------------------------
extern "C" void kernel_launch(void* const* d_in, const int* in_sizes, int n_in,
                              void* d_out, int out_size, void* d_ws, size_t ws_size,
                              hipStream_t stream) {
  const float* x    = (const float*)d_in[0];
  const float* Wqkv = (const float*)d_in[1];
  const float* bqkv = (const float*)d_in[2];
  const float* Wout = (const float*)d_in[3];
  const float* bout = (const float*)d_in[4];
  float* out = (float*)d_out;

  char* ws = (char*)d_ws;
  _Float16* xh  = (_Float16*)(ws);
  _Float16* wqt = (_Float16*)(ws + ((size_t)8 << 20));
  _Float16* wot = (_Float16*)(ws + ((size_t)14 << 20));
  _Float16* Qd  = (_Float16*)(ws + ((size_t)16 << 20));
  _Float16* Kd  = (_Float16*)(ws + ((size_t)24 << 20));
  _Float16* VTd = (_Float16*)(ws + ((size_t)32 << 20));

  const int nX = NTOK * CDIM;          // 4,194,304
  cvt_f16_kernel<<<nX / 256, 256, 0, stream>>>(x, xh, nX);
  cvt_t_f16_kernel<<<(CDIM * 3 * CDIM) / 256, 256, 0, stream>>>(Wqkv, wqt, CDIM, 3 * CDIM);
  cvt_t_f16_kernel<<<(CDIM * CDIM) / 256, 256, 0, stream>>>(Wout, wot, CDIM, CDIM);

  // 64 row-blocks x 48 col-blocks
  qkv_gemm_kernel<<<3072, 128, 0, stream>>>(xh, wqt, bqkv, Qd, Kd, VTd);

  // 16 heads x 256 q-tiles = 4096 waves / 4 per block; output into xh region
  attn_kernel<<<1024, 128, 0, stream>>>(Qd, Kd, VTd, xh);

  // 64 row-blocks x 16 col-blocks
  out_gemm_kernel<<<1024, 128, 0, stream>>>(xh, wot, bout, out);
}
